// ViT_68272800137232
// MI455X (gfx1250) — matrix operand order repaired and checked by reading, not compiled, not run on hardware
//
#include <hip/hip_runtime.h>
#include <hip/hip_bf16.h>

#define LAYERS 8
#define HEADS 16
#define DHEAD 64
#define DMODEL 1024
#define DFF 4096
#define BBATCH 4
#define SSEQ 512
#define ROWS (BBATCH * SSEQ)  // 2048

typedef __attribute__((ext_vector_type(16))) __bf16 v16bf;
typedef __attribute__((ext_vector_type(8)))  float  v8f;

__device__ __forceinline__ unsigned short f2bf(float f) {
  union { float f; unsigned u; } x; x.f = f;
  unsigned r = x.u + 0x7FFFu + ((x.u >> 16) & 1u);  // round-to-nearest-even
  return (unsigned short)(r >> 16);
}

union FragU { v16bf v; unsigned u[8]; };

// A fragment: 16x32 bf16, A stored row-major [m][k] with row stride lda.
// ISA layout: lane L: m=L%16, half=L/16; VGPR v holds K pair
// K = k0 + 16*(v>>2) + 2*(v&3) + 8*half  (+0,+1) -> two contiguous 16B chunks per lane.
__device__ __forceinline__ v16bf load_a_frag(const unsigned short* A, int row0,
                                             int lda, int k0, int lane) {
  int m = lane & 15, half = lane >> 4;
  FragU f;
#pragma unroll
  for (int v = 0; v < 8; ++v) {
    int k = k0 + ((v >> 2) << 4) + ((v & 3) << 1) + (half << 3);
    f.u[v] = *(const unsigned*)(A + (size_t)(row0 + m) * lda + k);
  }
  return f.v;
}

// B fragment: 32x16 bf16, B staged [n][k] row-major with row stride ldb.
// ISA layout: lane L: n=L%16, khalf=L/16; VGPR v holds K = k0 + 16*(v>>2) + 2*(v&3) + 8*khalf (+0,+1)
// -> two contiguous 16B chunks per lane (same K order as A).
__device__ __forceinline__ v16bf load_b_frag(const unsigned short* B, int n0,
                                             int ldb, int k0, int lane) {
  int n = lane & 15, khalf = lane >> 4;
  FragU f;
#pragma unroll
  for (int v = 0; v < 8; ++v) {
    int k = k0 + ((v >> 2) << 4) + ((v & 3) << 1) + (khalf << 3);
    f.u[v] = *(const unsigned*)(B + (size_t)(n0 + n) * ldb + k);
  }
  return f.v;
}

#define WMMA_BF16(a, b, c) \
  __builtin_amdgcn_wmma_f32_16x16x32_bf16(false, (a), false, (b), (short)0, (c), false, false)

// CDNA5 async global->LDS copy (ASYNCcnt-tracked, no VGPR round-trip).
__device__ __forceinline__ void async_load_b128(unsigned lds_off, const void* gaddr) {
  asm volatile("global_load_async_to_lds_b128 %0, %1, off"
               :: "v"(lds_off), "v"(gaddr) : "memory");
}
__device__ __forceinline__ void wait_async0() {
  asm volatile("s_wait_asynccnt 0" ::: "memory");
}

// ---------------- elementwise / conversion kernels ----------------

__global__ void copy_f32_kernel(const float* __restrict__ in, float* __restrict__ out, int n) {
  int i = blockIdx.x * 256 + threadIdx.x;
  if (i < n) out[i] = in[i];
}

__global__ void conv_bf16_kernel(const float* __restrict__ in, unsigned short* __restrict__ out, int n) {
  int i = blockIdx.x * 256 + threadIdx.x;
  if (i < n) out[i] = f2bf(in[i]);
}

// out[n*K + k] = bf16(in[k*N + n])   (stage B as [n][k])
__global__ void convT_bf16_kernel(const float* __restrict__ in, unsigned short* __restrict__ out,
                                  int K, int N) {
  long long i = (long long)blockIdx.x * 256 + threadIdx.x;
  if (i < (long long)K * N) {
    int n = (int)(i / K), k = (int)(i % K);
    out[i] = f2bf(in[(size_t)k * N + n]);
  }
}

// ---------------- layernorm (fp32 in -> bf16 out) ----------------

__global__ __launch_bounds__(256)
void ln_bf16_kernel(const float* __restrict__ hin, const float* __restrict__ w,
                    const float* __restrict__ b, unsigned short* __restrict__ out) {
  __shared__ float red[256];
  int row = blockIdx.x;
  int t = threadIdx.x;
  const float* xr = hin + (size_t)row * DMODEL;
  float vals[4];
  float s = 0.f;
#pragma unroll
  for (int i = 0; i < 4; ++i) { vals[i] = xr[t + i * 256]; s += vals[i]; }
  red[t] = s; __syncthreads();
  for (int o = 128; o > 0; o >>= 1) { if (t < o) red[t] += red[t + o]; __syncthreads(); }
  float mean = red[0] * (1.f / DMODEL);
  __syncthreads();
  float s2 = 0.f;
#pragma unroll
  for (int i = 0; i < 4; ++i) { float d = vals[i] - mean; s2 += d * d; }
  red[t] = s2; __syncthreads();
  for (int o = 128; o > 0; o >>= 1) { if (t < o) red[t] += red[t + o]; __syncthreads(); }
  float inv = rsqrtf(red[0] * (1.f / DMODEL) + 1e-5f);
  unsigned short* orow = out + (size_t)row * DMODEL;
#pragma unroll
  for (int i = 0; i < 4; ++i) {
    int c = t + i * 256;
    orow[c] = f2bf((vals[i] - mean) * inv * w[c] + b[c]);
  }
}

// ---------------- fused per-head QKV projection ----------------
// Each wave: 16 rows x one head. Q -> [m][h*64+e]; K -> Ks[b,h][s][e]; V -> Vt[b,h][e][s].

__global__ __launch_bounds__(256)
void qkv_kernel(const unsigned short* __restrict__ Y,
                const unsigned short* __restrict__ Bq,
                const unsigned short* __restrict__ Bk,
                const unsigned short* __restrict__ Bv,
                const float* __restrict__ qb, const float* __restrict__ kb,
                const float* __restrict__ vb,
                unsigned short* __restrict__ Qs, unsigned short* __restrict__ Ks,
                unsigned short* __restrict__ Vt) {
  int wid  = blockIdx.x * 8 + (threadIdx.x >> 5);
  int lane = threadIdx.x & 31;
  int mt = wid >> 4;     // 0..127 row tile
  int hh = wid & 15;     // head
  const unsigned short* Yh = Y + hh * DHEAD;
  const unsigned short* bq = Bq + hh * DHEAD * DHEAD;
  const unsigned short* bk = Bk + hh * DHEAD * DHEAD;
  const unsigned short* bv = Bv + hh * DHEAD * DHEAD;
  v8f qa[4] = {}; v8f ka[4] = {}; v8f va[4] = {};
#pragma unroll
  for (int ks = 0; ks < 2; ++ks) {
    v16bf a = load_a_frag(Yh, mt * 16, DMODEL, ks * 32, lane);
#pragma unroll
    for (int j = 0; j < 4; ++j) {
      qa[j] = WMMA_BF16(a, load_b_frag(bq, j * 16, DHEAD, ks * 32, lane), qa[j]);
      ka[j] = WMMA_BF16(a, load_b_frag(bk, j * 16, DHEAD, ks * 32, lane), ka[j]);
      va[j] = WMMA_BF16(a, load_b_frag(bv, j * 16, DHEAD, ks * 32, lane), va[j]);
    }
  }
  int nlo = lane & 15, mh = (lane >> 4) << 3;
#pragma unroll
  for (int j = 0; j < 4; ++j) {
    int e = j * 16 + nlo;
    float qbv = qb[hh * DHEAD + e], kbv = kb[hh * DHEAD + e], vbv = vb[hh * DHEAD + e];
#pragma unroll
    for (int v = 0; v < 8; ++v) {
      int m = mt * 16 + mh + v;
      int bidx = m >> 9, sl = m & 511;
      Qs[(size_t)m * DMODEL + hh * DHEAD + e]                    = f2bf(qa[j][v] + qbv);
      Ks[(((size_t)(bidx * HEADS + hh) * SSEQ + sl) << 6) + e]   = f2bf(ka[j][v] + kbv);
      Vt[(((size_t)(bidx * HEADS + hh) * DHEAD + e) << 9) + sl]  = f2bf(va[j][v] + vbv);
    }
  }
}

// ---------------- fused attention: scores -> softmax -> PV -> residual ----------------
// One wave per (b, h, 16-query tile). Scores kept in LDS (16x512 fp32), probs bf16 in LDS.

__global__ __launch_bounds__(32)
void attn_kernel(const unsigned short* __restrict__ Qs,
                 const unsigned short* __restrict__ Ks,
                 const unsigned short* __restrict__ Vt,
                 float* __restrict__ h) {
  __shared__ float sc[16 * SSEQ];
  __shared__ unsigned short pb[16 * SSEQ];
  int lane = threadIdx.x;
  int mt = blockIdx.x & 31;
  int bh = blockIdx.x >> 5;          // b*16 + head
  int bidx = bh >> 4, hh = bh & 15;
  const unsigned short* KsB = Ks + (size_t)bh * SSEQ * DHEAD;  // [s][e]
  const unsigned short* VtB = Vt + (size_t)bh * DHEAD * SSEQ;  // [e][s]
  int row0 = bidx * SSEQ + mt * 16;
  v16bf qa0 = load_a_frag(Qs + hh * DHEAD, row0, DMODEL, 0, lane);
  v16bf qa1 = load_a_frag(Qs + hh * DHEAD, row0, DMODEL, 32, lane);
  int nlo = lane & 15, mh = (lane >> 4) << 3;
  // Phase 1: scores = Q K^T * 1/sqrt(64)
  for (int nt = 0; nt < 32; ++nt) {
    v8f acc = {};
    acc = WMMA_BF16(qa0, load_b_frag(KsB, nt * 16, DHEAD, 0, lane), acc);
    acc = WMMA_BF16(qa1, load_b_frag(KsB, nt * 16, DHEAD, 32, lane), acc);
#pragma unroll
    for (int v = 0; v < 8; ++v) sc[(mh + v) * SSEQ + nt * 16 + nlo] = acc[v] * 0.125f;
  }
  __syncthreads();
  // Phase 2: softmax; 16 rows x 2 column halves across the wave32
  {
    int r = lane & 15;
    int c0 = (lane >> 4) * 256;
    float* srow = sc + r * SSEQ + c0;
    float mx = -3.0e38f;
    for (int c = 0; c < 256; ++c) mx = fmaxf(mx, srow[c]);
    mx = fmaxf(mx, __shfl_xor(mx, 16, 32));
    float sum = 0.f;
    for (int c = 0; c < 256; ++c) { float e = __expf(srow[c] - mx); sum += e; srow[c] = e; }
    sum += __shfl_xor(sum, 16, 32);
    float inv = 1.f / sum;
    unsigned short* prow = pb + r * SSEQ + c0;
    for (int c = 0; c < 256; ++c) prow[c] = f2bf(srow[c] * inv);
  }
  __syncthreads();
  // Phase 3: O = P @ V
  v8f o[4] = {};
  for (int kt = 0; kt < 16; ++kt) {
    v16bf a = load_a_frag(pb, 0, SSEQ, kt * 32, lane);
#pragma unroll
    for (int j = 0; j < 4; ++j)
      o[j] = WMMA_BF16(a, load_b_frag(VtB, j * 16, SSEQ, kt * 32, lane), o[j]);
  }
  // Phase 4: residual add into hidden state (each block owns a disjoint region)
#pragma unroll
  for (int j = 0; j < 4; ++j) {
    int e = j * 16 + nlo;
#pragma unroll
    for (int v = 0; v < 8; ++v) {
      int m = row0 + mh + v;
      h[(size_t)m * DMODEL + hh * DHEAD + e] += o[j][v];
    }
  }
}

// ---------------- bf16 WMMA GEMM, LDS-staged B panel via async-to-LDS ----------------
// C[M,N] = A[M,K](bf16 row-major) * B (staged Bt=[n][k] bf16) + bias
// Block: 256 threads = 8 waves; block tile 256(m) x 64(n); wave tile 32x64.
// B panel (64 x 128k, padded rows of 136) staged in LDS with
// global_load_async_to_lds_b128, consumed via ds_load_b128 fragments.
// flags bit0: exact GELU; bit1: accumulate fp32 into outF (else bf16 to outB).

#define BLROW 136  // 68 dwords: fragment lanes hit banks 4n..4n+3, conflict-free

__global__ __launch_bounds__(256)
void gemm_bf16_kernel(const unsigned short* __restrict__ A,
                      const unsigned short* __restrict__ Bt,
                      const float* __restrict__ bias,
                      unsigned short* __restrict__ outB,
                      float* __restrict__ outF,
                      int M, int N, int K, int flags) {
  __shared__ unsigned short Bl[64 * BLROW];
  int lane = threadIdx.x & 31;
  int wv   = threadIdx.x >> 5;           // 0..7
  int nStr = N >> 6;
  int bn = blockIdx.x % nStr;            // n-strip (64 cols)
  int bm = blockIdx.x / nStr;            // m-block (256 rows)
  int n0 = bn * 64;
  int m0 = bm * 256 + wv * 32;           // this wave's 32 rows
  unsigned ldsbase = (unsigned)(size_t)(&Bl[0]);

  v8f acc0[4] = {}; v8f acc1[4] = {};
  for (int k0 = 0; k0 < K; k0 += 128) {
    // cooperative async stage of B[n0..n0+63][k0..k0+127]: 1024 x 16B segments
#pragma unroll
    for (int i = 0; i < 4; ++i) {
      int s = threadIdx.x + i * 256;
      int n = s >> 4, seg = s & 15;
      const void* g = (const void*)(Bt + (size_t)(n0 + n) * K + k0 + seg * 8);
      async_load_b128(ldsbase + (unsigned)((n * BLROW + seg * 8) * 2), g);
    }
    wait_async0();
    __syncthreads();
#pragma unroll
    for (int kk = 0; kk < 128; kk += 32) {
      v16bf a0 = load_a_frag(A, m0,      K, k0 + kk, lane);
      v16bf a1 = load_a_frag(A, m0 + 16, K, k0 + kk, lane);
#pragma unroll
      for (int j = 0; j < 4; ++j) {
        v16bf b = load_b_frag(Bl, j * 16, BLROW, kk, lane);
        acc0[j] = WMMA_BF16(a0, b, acc0[j]);
        acc1[j] = WMMA_BF16(a1, b, acc1[j]);
      }
    }
    __syncthreads();  // protect LDS panel before next chunk overwrites it
  }
  int nlo = lane & 15, mh = (lane >> 4) << 3;
#pragma unroll
  for (int j = 0; j < 4; ++j) {
    int n = n0 + j * 16 + nlo;
    float bv = bias ? bias[n] : 0.f;
#pragma unroll
    for (int r = 0; r < 2; ++r) {
      const v8f& a = r ? acc1[j] : acc0[j];
#pragma unroll
      for (int v = 0; v < 8; ++v) {
        int m = m0 + r * 16 + mh + v;
        float val = a[v] + bv;
        if (flags & 1) val = 0.5f * val * (1.f + erff(val * 0.70710678118f));
        if (flags & 2) outF[(size_t)m * N + n] += val;
        else           outB[(size_t)m * N + n] = f2bf(val);
      }
    }
  }
}

// ---------------- host orchestration ----------------

extern "C" void kernel_launch(void* const* d_in, const int* in_sizes, int n_in,
                              void* d_out, int out_size, void* d_ws, size_t ws_size,
                              hipStream_t stream) {
  (void)in_sizes; (void)n_in; (void)out_size; (void)ws_size;
  const float* x    = (const float*)d_in[0];
  const float* ln1w = (const float*)d_in[1];
  const float* ln1b = (const float*)d_in[2];
  const float* qw   = (const float*)d_in[3];
  const float* qb   = (const float*)d_in[4];
  const float* kw   = (const float*)d_in[5];
  const float* kb   = (const float*)d_in[6];
  const float* vw   = (const float*)d_in[7];
  const float* vb   = (const float*)d_in[8];
  const float* ln2w = (const float*)d_in[9];
  const float* ln2b = (const float*)d_in[10];
  const float* w1   = (const float*)d_in[11];
  const float* b1   = (const float*)d_in[12];
  const float* w2   = (const float*)d_in[13];
  const float* b2   = (const float*)d_in[14];
  float* h = (float*)d_out;  // running residual stream == final output

  char* ws = (char*)d_ws;
  size_t off = 0;
  auto take = [&](size_t bytes) -> void* {
    void* p = ws + off; off += (bytes + 255) & ~(size_t)255; return p;
  };
  unsigned short* Y   = (unsigned short*)take((size_t)ROWS * DMODEL * 2);   // LN1/LN2 out (bf16)
  unsigned short* Qs  = (unsigned short*)take((size_t)ROWS * DMODEL * 2);
  unsigned short* Ks  = (unsigned short*)take((size_t)ROWS * DMODEL * 2);   // [b,h][s][e]
  unsigned short* Vt  = (unsigned short*)take((size_t)ROWS * DMODEL * 2);   // [b,h][e][s]
  unsigned short* Bq  = (unsigned short*)take((size_t)HEADS * DHEAD * DHEAD * 2);
  unsigned short* Bk  = (unsigned short*)take((size_t)HEADS * DHEAD * DHEAD * 2);
  unsigned short* Bv  = (unsigned short*)take((size_t)HEADS * DHEAD * DHEAD * 2);
  unsigned short* W1t = (unsigned short*)take((size_t)DMODEL * DFF * 2);    // [n][k]
  unsigned short* W2t = (unsigned short*)take((size_t)DFF * DMODEL * 2);    // [n][k]
  unsigned short* U   = (unsigned short*)take((size_t)ROWS * DFF * 2);      // GELU output

  {
    int n = ROWS * DMODEL;
    copy_f32_kernel<<<(n + 255) / 256, 256, 0, stream>>>(x, h, n);
  }

  const int QKVW = HEADS * DHEAD * DHEAD;  // 65536 per layer per tensor
  for (int l = 0; l < LAYERS; ++l) {
    ln_bf16_kernel<<<ROWS, 256, 0, stream>>>(h, ln1w + l * DMODEL, ln1b + l * DMODEL, Y);
    conv_bf16_kernel<<<QKVW / 256, 256, 0, stream>>>(qw + (size_t)l * QKVW, Bq, QKVW);
    conv_bf16_kernel<<<QKVW / 256, 256, 0, stream>>>(kw + (size_t)l * QKVW, Bk, QKVW);
    conv_bf16_kernel<<<QKVW / 256, 256, 0, stream>>>(vw + (size_t)l * QKVW, Bv, QKVW);
    qkv_kernel<<<(ROWS / 16) * HEADS / 8, 256, 0, stream>>>(
        Y, Bq, Bk, Bv,
        qb + (size_t)l * HEADS * DHEAD, kb + (size_t)l * HEADS * DHEAD,
        vb + (size_t)l * HEADS * DHEAD, Qs, Ks, Vt);
    attn_kernel<<<BBATCH * HEADS * (SSEQ / 16), 32, 0, stream>>>(Qs, Ks, Vt, h);
    ln_bf16_kernel<<<ROWS, 256, 0, stream>>>(h, ln2w + l * DMODEL, ln2b + l * DMODEL, Y);
    convT_bf16_kernel<<<(DMODEL * DFF + 255) / 256, 256, 0, stream>>>(
        w1 + (size_t)l * DMODEL * DFF, W1t, DMODEL, DFF);
    gemm_bf16_kernel<<<(ROWS / 256) * (DFF / 64), 256, 0, stream>>>(
        Y, W1t, b1 + (size_t)l * DFF, U, nullptr, ROWS, DFF, DMODEL, /*gelu*/1);
    convT_bf16_kernel<<<(DMODEL * DFF + 255) / 256, 256, 0, stream>>>(
        w2 + (size_t)l * DMODEL * DFF, W2t, DFF, DMODEL);
    gemm_bf16_kernel<<<(ROWS / 256) * (DMODEL / 64), 256, 0, stream>>>(
        U, W2t, b2 + (size_t)l * DMODEL, nullptr, h, ROWS, DMODEL, DFF, /*acc fp32*/2);
  }
}
